// GaussianSplattingSimple_86346022519338
// MI455X (gfx1250) — matrix operand order repaired and checked by reading, not compiled, not run on hardware
//
#include <hip/hip_runtime.h>
#include <hip/hip_bf16.h>

typedef __attribute__((ext_vector_type(16))) _Float16 v16h;
typedef __attribute__((ext_vector_type(8)))  _Float16 v8h;
typedef __attribute__((ext_vector_type(8)))  float    v8f;

#define DEV __device__ __forceinline__

// ---------------- gaussian dictionary constants ----------------
__device__ __constant__ float GS_CHO1[9] = {0.f, 0.41f, 0.62f, 0.98f, 1.13f, 1.29f, 1.64f, 1.85f, 2.36f};
__device__ __constant__ float GS_CHO2[9] = {-0.86f, -0.36f, -0.16f, 0.19f, 0.34f, 0.49f, 0.84f, 1.04f, 1.54f};
__device__ __constant__ float GS_CHO3[9] = {0.f, 0.33f, 0.53f, 0.88f, 1.03f, 1.18f, 1.53f, 1.73f, 2.23f};

DEV void gs_gau3(int n, float& a, float& b, float& c) {
    if (n >= 729) { a = b = c = 0.f; return; }
    int i1 = n / 81, r = n % 81;
    int i2 = r / 9, i3 = r % 9;
    a = GS_CHO1[i1]; b = GS_CHO2[i2]; c = GS_CHO3[i3];
}

// ---------------- weight packing into WMMA-swizzled B layout ----------------
// Bp[((kt*ntiles + nt)*32 + lane)*16 + e] = W[kt*32 + (lane>>4)*16 + e, nt*16 + (lane&15)]
__global__ __launch_bounds__(256) void gs_pack_fc(const float* __restrict__ W,
                                                  _Float16* __restrict__ Bp,
                                                  int K, int N, int Npad) {
    int idx = blockIdx.x * 256 + threadIdx.x;
    int total = K * Npad;
    if (idx >= total) return;
    int e = idx & 15;
    int lane = (idx >> 4) & 31;
    int t = idx >> 9;
    int ntiles = Npad >> 4;
    int nt = t % ntiles;
    int kt = t / ntiles;
    int k = kt * 32 + ((lane >> 4) << 4) + e;
    int n = nt * 16 + (lane & 15);
    float v = (n < N) ? W[(size_t)k * N + n] : 0.f;
    Bp[idx] = (_Float16)v;
}

// Conv OIHW (O,C,3,3) -> B [K=9C, N=O], k = (dy*3+dx)*C + c
__global__ __launch_bounds__(256) void gs_pack_conv(const float* __restrict__ W,
                                                    _Float16* __restrict__ Bp,
                                                    int C, int O) {
    int idx = blockIdx.x * 256 + threadIdx.x;
    int K = 9 * C;
    int total = K * O;
    if (idx >= total) return;
    int e = idx & 15;
    int lane = (idx >> 4) & 31;
    int t = idx >> 9;
    int ntiles = O >> 4;
    int nt = t % ntiles;
    int kt = t / ntiles;
    int k = kt * 32 + ((lane >> 4) << 4) + e;
    int n = nt * 16 + (lane & 15);
    int c = k % C;
    int dd = k / C;
    int dy = dd / 3, dx = dd % 3;
    float v = W[(((size_t)n * C + c) * 3 + dy) * 3 + dx];
    Bp[idx] = (_Float16)v;
}

// ---------------- NCHW f32 -> padded NHWC f16 ----------------
__global__ __launch_bounds__(256) void gs_pad_nchw(const float* __restrict__ in,
                                                   _Float16* __restrict__ out,
                                                   int B, int C, int H, int W) {
    int idx = blockIdx.x * 256 + threadIdx.x;
    int Hp = H + 2, Wp = W + 2;
    int total = B * Hp * Wp * C;
    if (idx >= total) return;
    int c = idx % C;
    int t = idx / C;
    int x = t % Wp; t /= Wp;
    int y = t % Hp;
    int b = t / Hp;
    float v = 0.f;
    if (y >= 1 && y <= H && x >= 1 && x <= W)
        v = in[(((size_t)b * C + c) * H + (y - 1)) * W + (x - 1)];
    out[idx] = (_Float16)v;
}

// NHWC f16 -> padded NHWC f16
__global__ __launch_bounds__(256) void gs_pad_nhwc_h(const _Float16* __restrict__ in,
                                                     _Float16* __restrict__ out,
                                                     int B, int H, int W, int C) {
    int idx = blockIdx.x * 256 + threadIdx.x;
    int Hp = H + 2, Wp = W + 2;
    int total = B * Hp * Wp * C;
    if (idx >= total) return;
    int c = idx % C;
    int t = idx / C;
    int x = t % Wp; t /= Wp;
    int y = t % Hp;
    int b = t / Hp;
    _Float16 v = (_Float16)0.f;
    if (y >= 1 && y <= H && x >= 1 && x <= W)
        v = in[(((size_t)b * H + (y - 1)) * W + (x - 1)) * C + c];
    out[idx] = v;
}

// ---------------- im2col (vectorized 8 halves) ----------------
// A[row=(b,y,x), kk=(dy*3+dx)*C + c] = pad[b, y+dy, x+dx, c]
__global__ __launch_bounds__(256) void gs_im2col(const _Float16* __restrict__ pad,
                                                 _Float16* __restrict__ A,
                                                 int B, int H, int W, int C) {
    int idx = blockIdx.x * 256 + threadIdx.x;
    int C8 = C >> 3;
    int total = B * H * W * 9 * C8;
    if (idx >= total) return;
    int c8 = idx % C8;
    int t = idx / C8;
    int dd = t % 9;
    int row = t / 9;
    int dy = dd / 3, dx = dd % 3;
    int hw = H * W;
    int b = row / hw;
    int r = row % hw;
    int y = r / W, x = r % W;
    const uint4* src = (const uint4*)(pad + ((((size_t)b * (H + 2) + y + dy) * (W + 2) + x + dx) * C + c8 * 8));
    uint4* dst = (uint4*)(A + ((size_t)row * 9 * C + dd * C + c8 * 8));
    *dst = *src;
}

// ---------------- generic WMMA GEMM ----------------
// A: [M,K] f16 row-major.  Bp: swizzled (see gs_pack_fc).  C: [M,Npad] f32.
// act: 0 none, 1 relu, 2 leaky(0.1).  Ch optional f16 duplicate.
// Wave tile: 32(M) x 64(N); block = 8 waves = 256(M) x 64(N).
DEV void gs_store_frag(v8f acc, int colj, int mrow0, int Npad,
                       const float* bias, float* C, _Float16* Ch, int act) {
    float bv = bias ? bias[colj] : 0.f;
#pragma unroll
    for (int r = 0; r < 8; ++r) {
        float v = acc[r] + bv;
        if (act == 1) v = fmaxf(v, 0.f);
        else if (act == 2) v = (v >= 0.f) ? v : 0.1f * v;
        size_t o = (size_t)(mrow0 + r) * Npad + colj;
        C[o] = v;
        if (Ch) Ch[o] = (_Float16)v;
    }
}

DEV v16h gs_load_afrag(const _Float16* p) {
    v8h lo = *(const v8h*)(p);
    v8h hi = *(const v8h*)(p + 8);
    return __builtin_shufflevector(lo, hi, 0, 1, 2, 3, 4, 5, 6, 7, 8, 9, 10, 11, 12, 13, 14, 15);
}

__global__ __launch_bounds__(256)
void gs_gemm_wmma(const _Float16* __restrict__ A, const _Float16* __restrict__ Bp,
                  const float* __restrict__ bias, float* __restrict__ C,
                  _Float16* __restrict__ Ch, int M, int Npad, int K, int act) {
    const int lane = threadIdx.x & 31;
    const int wv = threadIdx.x >> 5;
    const int mBase = (blockIdx.y * 8 + wv) * 32;
    const int nBase = blockIdx.x * 64;
    const int hh = lane >> 4;
    const int ntiles = Npad >> 4;
    const int nt0 = nBase >> 4;

    v8f a0c0{}, a0c1{}, a0c2{}, a0c3{};
    v8f a1c0{}, a1c1{}, a1c2{}, a1c3{};

    const _Float16* aptr0 = A + (size_t)(mBase + (lane & 15)) * K + hh * 16;
    const _Float16* aptr1 = aptr0 + (size_t)16 * K;
    const _Float16* bbase = Bp + ((size_t)nt0 * 32 + lane) * 16;
    const size_t bkstride = (size_t)ntiles * 512;  // halves per k-tile

    const int KT = K >> 5;
    for (int kt = 0; kt < KT; ++kt) {
        v16h a0 = gs_load_afrag(aptr0);
        v16h a1 = gs_load_afrag(aptr1);
        aptr0 += 32;
        aptr1 += 32;
        const _Float16* bb = bbase + (size_t)kt * bkstride;
        v16h b0 = *(const v16h*)(bb);
        v16h b1 = *(const v16h*)(bb + 512);
        v16h b2 = *(const v16h*)(bb + 1024);
        v16h b3 = *(const v16h*)(bb + 1536);
        a0c0 = __builtin_amdgcn_wmma_f32_16x16x32_f16(false, a0, false, b0, (short)0, a0c0, false, false);
        a1c0 = __builtin_amdgcn_wmma_f32_16x16x32_f16(false, a1, false, b0, (short)0, a1c0, false, false);
        a0c1 = __builtin_amdgcn_wmma_f32_16x16x32_f16(false, a0, false, b1, (short)0, a0c1, false, false);
        a1c1 = __builtin_amdgcn_wmma_f32_16x16x32_f16(false, a1, false, b1, (short)0, a1c1, false, false);
        a0c2 = __builtin_amdgcn_wmma_f32_16x16x32_f16(false, a0, false, b2, (short)0, a0c2, false, false);
        a1c2 = __builtin_amdgcn_wmma_f32_16x16x32_f16(false, a1, false, b2, (short)0, a1c2, false, false);
        a0c3 = __builtin_amdgcn_wmma_f32_16x16x32_f16(false, a0, false, b3, (short)0, a0c3, false, false);
        a1c3 = __builtin_amdgcn_wmma_f32_16x16x32_f16(false, a1, false, b3, (short)0, a1c3, false, false);
    }

    const int c0 = lane & 15;
    const int mrow0 = mBase + hh * 8;
    const int mrow1 = mBase + 16 + hh * 8;
    gs_store_frag(a0c0, nBase + 0 + c0, mrow0, Npad, bias, C, Ch, act);
    gs_store_frag(a0c1, nBase + 16 + c0, mrow0, Npad, bias, C, Ch, act);
    gs_store_frag(a0c2, nBase + 32 + c0, mrow0, Npad, bias, C, Ch, act);
    gs_store_frag(a0c3, nBase + 48 + c0, mrow0, Npad, bias, C, Ch, act);
    gs_store_frag(a1c0, nBase + 0 + c0, mrow1, Npad, bias, C, Ch, act);
    gs_store_frag(a1c1, nBase + 16 + c0, mrow1, Npad, bias, C, Ch, act);
    gs_store_frag(a1c2, nBase + 32 + c0, mrow1, Npad, bias, C, Ch, act);
    gs_store_frag(a1c3, nBase + 48 + c0, mrow1, Npad, bias, C, Ch, act);
}

// ---------------- pixel_unshuffle(2) + f16 convert ----------------
// in: conv1 out [B*64*64, 256] f32 (leaky applied). out: [B*32*32, 1024] f16
__global__ __launch_bounds__(256) void gs_unshuffle(const float* __restrict__ y,
                                                    _Float16* __restrict__ ff) {
    int idx = blockIdx.x * 256 + threadIdx.x;
    if (idx >= 2048 * 1024) return;
    int cc = idx & 1023;
    int row = idx >> 10;
    int b = row >> 10;
    int r = row & 1023;
    int y2 = r >> 5, x2 = r & 31;
    int c = cc >> 2;
    int s = cc & 3;
    int sy = s >> 1, sx = s & 1;
    int in_row = b * 4096 + (2 * y2 + sy) * 64 + (2 * x2 + sx);
    ff[idx] = (_Float16)y[(size_t)in_row * 256 + c];
}

// ---------------- dictionary MLP ----------------
__global__ __launch_bounds__(256) void gs_dictA(const float* __restrict__ w1,
                                                const float* __restrict__ b1,
                                                float* __restrict__ d) {
    int idx = blockIdx.x * 256 + threadIdx.x;
    if (idx >= 730 * 256) return;
    int j = idx & 255;
    int n = idx >> 8;
    float g0, g1, g2;
    gs_gau3(n, g0, g1, g2);
    float v = g0 * w1[j] + g1 * w1[256 + j] + g2 * w1[512 + j] + b1[j];
    d[idx] = fmaxf(v, 0.f);
}

// embed^T [512, 768] written directly in WMMA-swizzled layout (zeros for n>=730)
__global__ __launch_bounds__(256) void gs_dictB_pack(const float* __restrict__ d,
                                                     const float* __restrict__ w2,
                                                     const float* __restrict__ b2,
                                                     _Float16* __restrict__ Bp) {
    int idx = blockIdx.x * 256 + threadIdx.x;
    if (idx >= 512 * 768) return;
    int e = idx & 15;
    int lane = (idx >> 4) & 31;
    int t = idx >> 9;
    int nt = t % 48;   // 768/16
    int kt = t / 48;
    int k = kt * 32 + ((lane >> 4) << 4) + e;
    int n = nt * 16 + (lane & 15);
    float s = 0.f;
    if (n < 730) {
        s = b2[k];
        const float* dr = d + (size_t)n * 256;
        for (int m = 0; m < 256; ++m) s += dr[m] * w2[(size_t)m * 512 + k];
    }
    Bp[idx] = (_Float16)s;
}

// ---------------- small scalar heads ----------------
__global__ __launch_bounds__(256) void gs_col3(const float* __restrict__ c2,
                                               const float* __restrict__ w3,
                                               const float* __restrict__ b3,
                                               float* __restrict__ colors) {
    int idx = blockIdx.x * 256 + threadIdx.x;
    if (idx >= 2048 * 3) return;
    int row = idx / 3, j = idx % 3;
    float s = b3[j];
    const float* cr = c2 + (size_t)row * 256;
    for (int k = 0; k < 256; ++k) s += cr[k] * w3[k * 3 + j];
    colors[idx] = s;
}

__global__ __launch_bounds__(256) void gs_off2(const float* __restrict__ o1,
                                               const float* __restrict__ w2,
                                               const float* __restrict__ b2,
                                               float* __restrict__ offs) {
    int idx = blockIdx.x * 256 + threadIdx.x;
    if (idx >= 2048 * 2) return;
    int row = idx / 2, j = idx % 2;
    float s = b2[j];
    const float* orow = o1 + (size_t)row * 256;
    for (int k = 0; k < 256; ++k) s += orow[k] * w2[k * 2 + j];
    offs[idx] = tanhf(s);
}

// ---------------- softmax over dictionary + weighted sum with gau_dict ----------------
// scores [2048, 768] (valid cols: 730). One wave per row.
__global__ __launch_bounds__(256) void gs_softmax_cov(const float* __restrict__ scores,
                                                      float* __restrict__ covp) {
    int row = blockIdx.x * 8 + (threadIdx.x >> 5);
    int lane = threadIdx.x & 31;
    const float* s = scores + (size_t)row * 768;
    float m = -1e30f;
    for (int j = lane; j < 730; j += 32) m = fmaxf(m, s[j]);
#pragma unroll
    for (int o = 16; o; o >>= 1) m = fmaxf(m, __shfl_xor(m, o));
    float sum = 0.f, t0 = 0.f, t1 = 0.f, t2 = 0.f;
    for (int j = lane; j < 730; j += 32) {
        float w = __expf(s[j] - m);
        float g0, g1, g2;
        gs_gau3(j, g0, g1, g2);
        sum += w; t0 += w * g0; t1 += w * g1; t2 += w * g2;
    }
#pragma unroll
    for (int o = 16; o; o >>= 1) {
        sum += __shfl_xor(sum, o);
        t0 += __shfl_xor(t0, o);
        t1 += __shfl_xor(t1, o);
        t2 += __shfl_xor(t2, o);
    }
    if (lane == 0) {
        float inv = 1.f / sum;
        covp[row * 3 + 0] = t0 * inv;
        covp[row * 3 + 1] = t1 * inv;
        covp[row * 3 + 2] = t2 * inv;
    }
}

// ---------------- per-gaussian parameter prep ----------------
__global__ __launch_bounds__(256) void gs_prep(const float* __restrict__ covp,
                                               const float* __restrict__ offs,
                                               const float* __restrict__ cols,
                                               float* __restrict__ gp) {
    int n = blockIdx.x * 256 + threadIdx.x;
    if (n >= 2048) return;
    int idx = n & 1023;
    int i = idx >> 5, j = idx & 31;
    const float ry = 1.f / 32.f;
    float by = -1.f + ry * (2 * i + 1);
    float bx = -1.f + ry * (2 * j + 1);
    float cy = by + offs[n * 2 + 0] * (2.f / 32.f);
    float cx = bx + offs[n * 2 + 1] * (2.f / 32.f);
    const float scale = 2.f;  // out_h / H = 128/64
    float c1 = covp[n * 3 + 0] * scale + 1e-4f;
    float c2 = covp[n * 3 + 1] * scale;
    float c3 = covp[n * 3 + 2] * scale + 1e-4f;
    float sxx = c1 * c1, sxy = c1 * c2, syy = c2 * c2 + c3 * c3;
    float det = sxx * syy - sxy * sxy + 1e-6f;
    float inv = 1.f / det;
    float* g = gp + (size_t)n * 8;
    g[0] = cy; g[1] = cx;
    g[2] = syy * inv;   // A
    g[3] = sxx * inv;   // B
    g[4] = sxy * inv;   // C
    g[5] = cols[n * 3 + 0];
    g[6] = cols[n * 3 + 1];
    g[7] = cols[n * 3 + 2];
}

// ---------------- render: splat + keys-cubic upsample ----------------
DEV float gs_keys(float x) {
    x = fabsf(x);
    if (x <= 1.f) return ((1.5f * x - 2.5f) * x) * x + 1.f;
    if (x < 2.f) return ((-0.5f * x + 2.5f) * x - 4.f) * x + 2.f;
    return 0.f;
}

DEV void gs_cubic_w(int p, float* w, int* idx) {
    float u = (p + 0.5f) * 0.5f - 0.5f;
    int i0 = (int)floorf(u);
    float sum = 0.f;
#pragma unroll
    for (int t = 0; t < 4; ++t) {
        int k = i0 - 1 + t;
        float ww = gs_keys(u - (float)k);
        if (k < 0 || k > 63) { k = 0; ww = 0.f; }
        idx[t] = k;
        w[t] = ww;
        sum += ww;
    }
    float inv = 1.f / sum;
#pragma unroll
    for (int t = 0; t < 4; ++t) w[t] *= inv;
}

__global__ __launch_bounds__(256)
void gs_render(const float* __restrict__ gp, const float* __restrict__ lr,
               float* __restrict__ out) {
    __shared__ float sg[256 * 8];
    const int b = blockIdx.y;
    const int tid = threadIdx.x;
    const int pix = blockIdx.x * 256 + tid;
    const int py = pix >> 7, px = pix & 127;
    const float y = -1.f + 2.f * (float)py / 127.f;
    const float x = -1.f + 2.f * (float)px / 127.f;
    float n0 = 0.f, n1 = 0.f, n2 = 0.f, den = 0.f;

    for (int c0 = 0; c0 < 1024; c0 += 256) {
        __syncthreads();
        const float4* src = (const float4*)(gp + (size_t)(b * 1024 + c0 + tid) * 8);
        ((float4*)sg)[tid * 2 + 0] = src[0];
        ((float4*)sg)[tid * 2 + 1] = src[1];
        __syncthreads();
#pragma unroll 4
        for (int i = 0; i < 256; ++i) {
            const float* g = sg + i * 8;
            float dy = y - g[0];
            float dx = x - g[1];
            float mah = dy * dy * g[2] + dx * dx * g[3] - 2.f * dy * dx * g[4];
            float w = __expf(-0.5f * mah);
            den += w;
            n0 += w * g[5];
            n1 += w * g[6];
            n2 += w * g[7];
        }
    }
    float inv = 1.f / (den + 1e-6f);

    // bicubic upsample of lr (64x64 -> 128x128), keys a=-0.5, renormalized at edges
    float wy[4], wx[4];
    int iy[4], ix[4];
    gs_cubic_w(py, wy, iy);
    gs_cubic_w(px, wx, ix);
    float up[3] = {0.f, 0.f, 0.f};
#pragma unroll
    for (int a = 0; a < 4; ++a) {
#pragma unroll
        for (int t = 0; t < 4; ++t) {
            float w = wy[a] * wx[t];
            const float* base = lr + ((size_t)b * 3) * 4096 + iy[a] * 64 + ix[t];
            up[0] += w * base[0];
            up[1] += w * base[4096];
            up[2] += w * base[8192];
        }
    }
    size_t o = (((size_t)b * 3) * 128 + py) * 128 + px;
    out[o] = n0 * inv + up[0];
    out[o + 16384] = n1 * inv + up[1];
    out[o + 32768] = n2 * inv + up[2];
}

// ---------------- host launch ----------------
static inline int gs_blocks(long long n) { return (int)((n + 255) / 256); }

extern "C" void kernel_launch(void* const* d_in, const int* in_sizes, int n_in,
                              void* d_out, int out_size, void* d_ws, size_t ws_size,
                              hipStream_t stream) {
    const float* feat    = (const float*)d_in[0];
    const float* lr      = (const float*)d_in[1];
    const float* conv1_w = (const float*)d_in[2];
    const float* conv1_b = (const float*)d_in[3];
    const float* cov_w   = (const float*)d_in[4];
    const float* cov_b   = (const float*)d_in[5];
    const float* col_w1  = (const float*)d_in[6];
    const float* col_b1  = (const float*)d_in[7];
    const float* col_w2  = (const float*)d_in[8];
    const float* col_b2  = (const float*)d_in[9];
    const float* col_w3  = (const float*)d_in[10];
    const float* col_b3  = (const float*)d_in[11];
    const float* off_w1  = (const float*)d_in[12];
    const float* off_b1  = (const float*)d_in[13];
    const float* off_w2  = (const float*)d_in[14];
    const float* off_b2  = (const float*)d_in[15];
    const float* dict_w1 = (const float*)d_in[16];
    const float* dict_b1 = (const float*)d_in[17];
    const float* dict_w2 = (const float*)d_in[18];
    const float* dict_b2 = (const float*)d_in[19];
    float* out = (float*)d_out;

    char* p = (char*)d_ws;
    auto alloc = [&](size_t bytes) -> char* {
        char* r = p;
        p += (bytes + 255) & ~(size_t)255;
        return r;
    };

    _Float16* Wc1p   = (_Float16*)alloc((size_t)576 * 256 * 2);
    _Float16* Wcovp  = (_Float16*)alloc((size_t)9216 * 512 * 2);
    _Float16* W1p    = (_Float16*)alloc((size_t)1024 * 512 * 2);
    _Float16* W2p    = (_Float16*)alloc((size_t)512 * 256 * 2);
    _Float16* Wop    = (_Float16*)alloc((size_t)1024 * 256 * 2);
    _Float16* embTp  = (_Float16*)alloc((size_t)512 * 768 * 2);
    _Float16* fpad   = (_Float16*)alloc((size_t)2 * 66 * 66 * 64 * 2);
    _Float16* A1     = (_Float16*)alloc((size_t)8192 * 576 * 2);
    float*    c1out  = (float*)alloc((size_t)8192 * 256 * 4);
    _Float16* ffh    = (_Float16*)alloc((size_t)2048 * 1024 * 2);
    _Float16* ffpad  = (_Float16*)alloc((size_t)2 * 34 * 34 * 1024 * 2);
    _Float16* A2     = (_Float16*)alloc((size_t)2048 * 9216 * 2);
    float*    covf   = (float*)alloc((size_t)2048 * 512 * 4);
    _Float16* covfh  = (_Float16*)alloc((size_t)2048 * 512 * 2);
    float*    c1     = (float*)alloc((size_t)2048 * 512 * 4);
    _Float16* c1h    = (_Float16*)alloc((size_t)2048 * 512 * 2);
    float*    c2     = (float*)alloc((size_t)2048 * 256 * 4);
    float*    o1     = (float*)alloc((size_t)2048 * 256 * 4);
    float*    scores = (float*)alloc((size_t)2048 * 768 * 4);
    float*    dbuf   = (float*)alloc((size_t)730 * 256 * 4);
    float*    colors = (float*)alloc((size_t)2048 * 3 * 4);
    float*    offs   = (float*)alloc((size_t)2048 * 2 * 4);
    float*    covp   = (float*)alloc((size_t)2048 * 3 * 4);
    float*    gpar   = (float*)alloc((size_t)2048 * 8 * 4);
    (void)ws_size; (void)in_sizes; (void)n_in; (void)out_size;

    // --- weight packing ---
    gs_pack_conv<<<gs_blocks((long long)576 * 256), 256, 0, stream>>>(conv1_w, Wc1p, 64, 256);
    gs_pack_conv<<<gs_blocks((long long)9216 * 512), 256, 0, stream>>>(cov_w, Wcovp, 1024, 512);
    gs_pack_fc<<<gs_blocks((long long)1024 * 512), 256, 0, stream>>>(col_w1, W1p, 1024, 512, 512);
    gs_pack_fc<<<gs_blocks((long long)512 * 256), 256, 0, stream>>>(col_w2, W2p, 512, 256, 256);
    gs_pack_fc<<<gs_blocks((long long)1024 * 256), 256, 0, stream>>>(off_w1, Wop, 1024, 256, 256);

    // --- conv1 via im2col GEMM (M=8192,K=576,N=256), leaky 0.1 fused ---
    gs_pad_nchw<<<gs_blocks((long long)2 * 66 * 66 * 64), 256, 0, stream>>>(feat, fpad, 2, 64, 64, 64);
    gs_im2col<<<gs_blocks((long long)8192 * 9 * 8), 256, 0, stream>>>(fpad, A1, 2, 64, 64, 64);
    gs_gemm_wmma<<<dim3(256 / 64, 8192 / 256), 256, 0, stream>>>(A1, Wc1p, conv1_b, c1out, (_Float16*)nullptr,
                                                                 8192, 256, 576, 2);

    // --- pixel_unshuffle(2) -> feat_flat f16 [2048,1024] ---
    gs_unshuffle<<<gs_blocks((long long)2048 * 1024), 256, 0, stream>>>(c1out, ffh);

    // --- cov conv via im2col GEMM (M=2048,K=9216,N=512) ---
    gs_pad_nhwc_h<<<gs_blocks((long long)2 * 34 * 34 * 1024), 256, 0, stream>>>(ffh, ffpad, 2, 32, 32, 1024);
    gs_im2col<<<gs_blocks((long long)2048 * 9 * 128), 256, 0, stream>>>(ffpad, A2, 2, 32, 32, 1024);
    gs_gemm_wmma<<<dim3(512 / 64, 2048 / 256), 256, 0, stream>>>(A2, Wcovp, cov_b, covf, covfh,
                                                                 2048, 512, 9216, 0);

    // --- color MLP ---
    gs_gemm_wmma<<<dim3(512 / 64, 2048 / 256), 256, 0, stream>>>(ffh, W1p, col_b1, c1, c1h,
                                                                 2048, 512, 1024, 1);
    gs_gemm_wmma<<<dim3(256 / 64, 2048 / 256), 256, 0, stream>>>(c1h, W2p, col_b2, c2, (_Float16*)nullptr,
                                                                 2048, 256, 512, 1);
    gs_col3<<<gs_blocks((long long)2048 * 3), 256, 0, stream>>>(c2, col_w3, col_b3, colors);

    // --- offset MLP ---
    gs_gemm_wmma<<<dim3(256 / 64, 2048 / 256), 256, 0, stream>>>(ffh, Wop, off_b1, o1, (_Float16*)nullptr,
                                                                 2048, 256, 1024, 1);
    gs_off2<<<gs_blocks((long long)2048 * 2), 256, 0, stream>>>(o1, off_w2, off_b2, offs);

    // --- dictionary attention ---
    gs_dictA<<<gs_blocks((long long)730 * 256), 256, 0, stream>>>(dict_w1, dict_b1, dbuf);
    gs_dictB_pack<<<gs_blocks((long long)512 * 768), 256, 0, stream>>>(dbuf, dict_w2, dict_b2, embTp);
    gs_gemm_wmma<<<dim3(768 / 64, 2048 / 256), 256, 0, stream>>>(covfh, embTp, (const float*)nullptr, scores,
                                                                 (_Float16*)nullptr, 2048, 768, 512, 0);
    gs_softmax_cov<<<2048 / 8, 256, 0, stream>>>(scores, covp);

    // --- gaussian params + render ---
    gs_prep<<<gs_blocks(2048), 256, 0, stream>>>(covp, offs, colors, gpar);
    gs_render<<<dim3(128 * 128 / 256, 2), 256, 0, stream>>>(gpar, lr, out);
}
